// Decoder_87093346828868
// MI455X (gfx1250) — matrix operand order repaired and checked by reading, not compiled, not run on hardware
//
#include <hip/hip_runtime.h>

// ---------------------------------------------------------------------------
// Problem constants (from reference): B=16, H=8, I=512 (seq/key_dim), J=768
// ---------------------------------------------------------------------------
constexpr int B_ = 16;
constexpr int H_ = 8;
constexpr int I_ = 512;
constexpr int J_ = 768;

typedef __attribute__((ext_vector_type(16))) __bf16 v16bf;
typedef __attribute__((ext_vector_type(8)))  __bf16 v8bf;
typedef __attribute__((ext_vector_type(8)))  float  v8f;

#define AS1 __attribute__((address_space(1)))
#define AS3 __attribute__((address_space(3)))

// 128-bit int vector matching the async-LDS builtin's parameter type
typedef int gv4i __attribute__((vector_size(16)));

#if defined(__AMDGCN__) && __has_builtin(__builtin_amdgcn_global_load_async_to_lds_b128)
#define HAVE_ASYNC_LDS 1
#else
#define HAVE_ASYNC_LDS 0
#endif

__device__ __forceinline__ unsigned short f2bf(float f) {
    // round-to-nearest-even bf16
    unsigned int u = __float_as_uint(f);
    unsigned int r = u + 0x7FFFu + ((u >> 16) & 1u);
    return (unsigned short)(r >> 16);
}

__device__ __forceinline__ v16bf cat16(v8bf lo, v8bf hi) {
    return __builtin_shufflevector(lo, hi, 0,1,2,3,4,5,6,7,8,9,10,11,12,13,14,15);
}

#if HAVE_ASYNC_LDS
// 16B global -> LDS async copy (ASYNCcnt-tracked, no VGPR data round-trip).
// AS3 pointer built from low 32 bits of the flat address (== LDS offset per
// the gfx1250 aperture mapping, ISA 10.2).
__device__ __forceinline__ void async_copy16(const void* g, void* l) {
    __builtin_amdgcn_global_load_async_to_lds_b128(
        (AS1 gv4i*)(uintptr_t)g,
        (AS3 gv4i*)(unsigned)(uintptr_t)l, 0, 0);
}
__device__ __forceinline__ void wait_async0() {
#if __has_builtin(__builtin_amdgcn_s_wait_asynccnt)
    __builtin_amdgcn_s_wait_asynccnt(0);
#else
    asm volatile("s_wait_asynccnt 0x0" ::: "memory");
#endif
}
#endif

// ---------------------------------------------------------------------------
// Kernel 0: Q = bf16(x * W_q / sqrt(J)),  K = bf16(x * W_k)   (layout [b,h,i,j])
// ---------------------------------------------------------------------------
__global__ __launch_bounds__(256)
void prep_qk(const float* __restrict__ x, const float* __restrict__ Wq,
             const float* __restrict__ Wk,
             unsigned short* __restrict__ Qb, unsigned short* __restrict__ Kb)
{
    const float scale = 0.03608439182435161f; // 1/sqrt(768)
    const size_t total = (size_t)B_ * H_ * I_ * J_;
    for (size_t idx = (size_t)blockIdx.x * blockDim.x + threadIdx.x; idx < total;
         idx += (size_t)gridDim.x * blockDim.x) {
        size_t j = idx % J_;
        size_t r = idx / J_;
        size_t i = r % I_; r /= I_;
        size_t h = r % H_;
        size_t b = r / H_;
        float xv = x[(b * I_ + i) * J_ + j];
        size_t wi = (h * I_ + i) * J_ + j;
        Qb[idx] = f2bf(xv * Wq[wi] * scale);
        Kb[idx] = f2bf(xv * Wk[wi]);
    }
}

__global__ void zerof(float* __restrict__ p, int n) {
    int i = blockIdx.x * blockDim.x + threadIdx.x;
    if (i < n) p[i] = 0.f;
}

// ---------------------------------------------------------------------------
// Kernel 1: scores = Q Kt (bf16 WMMA), softmax over full rows in LDS,
//           accumulate attention column sums A[b,h,k] = sum_i attn[i,k].
// Block: 256 threads (8 waves), handles 64 rows x 512 cols, K=768 in chunks.
// ---------------------------------------------------------------------------
constexpr int TI = 64;        // rows per block
constexpr int KC = 128;       // K chunk staged in LDS
constexpr int QS_STR = 136;   // padded LDS row stride (shorts): 272B -> conflict-free
constexpr int KS_STR = 136;

__global__ __launch_bounds__(256)
void attn_kernel(const unsigned short* __restrict__ Qb,
                 const unsigned short* __restrict__ Kb,
                 float* __restrict__ A)
{
    __shared__ __align__(16) float          s [TI * 512];        // 131072 B
    __shared__ __align__(16) unsigned short qs[TI * QS_STR];     //  17408 B
    __shared__ __align__(16) unsigned short ks[512 * KS_STR];    // 139264 B
    __shared__ float zpart[TI * 4];
    __shared__ float rinv [TI];

    const int bid = blockIdx.x;
    const int rb = bid & 7;          // row-block within (b,h): 8 * 64 = 512 rows
    const int h  = (bid >> 3) & 7;
    const int b  = bid >> 6;
    const int t  = threadIdx.x;
    const int wave = t >> 5;
    const int lane = t & 31;
    const int lrow  = lane & 15;
    const int lhalf = lane >> 4;
    const int m_tile = wave & 3;     // 4 M tiles of 16 -> 64 rows
    const int n_half = wave >> 2;    // 2-way split of 32 N tiles

    const unsigned short* Qrow = Qb + ((size_t)(b * H_ + h) * I_ + (size_t)rb * TI) * J_;
    const unsigned short* Krow = Kb + ((size_t)(b * H_ + h) * I_) * J_;

    const v8f zero8 = {0.f,0.f,0.f,0.f,0.f,0.f,0.f,0.f};
    v8f acc[16];
#pragma unroll
    for (int n = 0; n < 16; ++n) acc[n] = zero8;

    for (int kc = 0; kc < J_; kc += KC) {
        // ---- stage Q tile (64 x 128 bf16) and K tile (512 x 128 bf16), 16B chunks
#if HAVE_ASYNC_LDS
        for (int e = t; e < (TI * KC) / 8; e += 256) {
            int r = e >> 4; int c = (e & 15) << 3;
            async_copy16(&Qrow[(size_t)r * J_ + kc + c], &qs[r * QS_STR + c]);
        }
        for (int e = t; e < (512 * KC) / 8; e += 256) {
            int r = e >> 4; int c = (e & 15) << 3;
            async_copy16(&Krow[(size_t)r * J_ + kc + c], &ks[r * KS_STR + c]);
        }
        wait_async0();
#else
        for (int e = t; e < (TI * KC) / 8; e += 256) {
            int r = e >> 4; int c = (e & 15) << 3;
            *reinterpret_cast<uint4*>(&qs[r * QS_STR + c]) =
                *reinterpret_cast<const uint4*>(&Qrow[(size_t)r * J_ + kc + c]);
        }
        for (int e = t; e < (512 * KC) / 8; e += 256) {
            int r = e >> 4; int c = (e & 15) << 3;
            *reinterpret_cast<uint4*>(&ks[r * KS_STR + c]) =
                *reinterpret_cast<const uint4*>(&Krow[(size_t)r * J_ + kc + c]);
        }
#endif
        __syncthreads();

        // ---- WMMA: for each k-step (K=32) and each of this wave's 16 N tiles
#pragma unroll
        for (int kk = 0; kk < KC / 32; ++kk) {
            // A fragment (16x32 bf16): lane holds row M=lrow;
            // elements 0..7 = K[half*8 .. +7], 8..15 = K[16+half*8 .. +7]
            const unsigned short* arow = &qs[(m_tile * 16 + lrow) * QS_STR + kk * 32];
            v16bf afrag = cat16(*reinterpret_cast<const v8bf*>(arow + lhalf * 8),
                                *reinterpret_cast<const v8bf*>(arow + 16 + lhalf * 8));
            // B fragment (32x16 bf16): lane holds column N=lrow (= key row),
            // elements 0..7 = K[half*8 .. +7], 8..15 = K[16+half*8 .. +7].  Double-buffered so the
            // ds_load for tile nt+1 issues before the wmma for tile nt waits.
            auto loadB = [&](int nt) -> v16bf {
                int n_tile = nt * 2 + n_half;
                const unsigned short* brow =
                    &ks[(n_tile * 16 + lrow) * KS_STR + kk * 32 + lhalf * 8];
                return cat16(*reinterpret_cast<const v8bf*>(brow),
                             *reinterpret_cast<const v8bf*>(brow + 16));
            };
            v16bf bcur = loadB(0);
#pragma unroll
            for (int nt = 0; nt < 16; ++nt) {
                v16bf bnxt = (nt < 15) ? loadB(nt + 1) : bcur;
                acc[nt] = __builtin_amdgcn_wmma_f32_16x16x32_bf16(
                    false, afrag, false, bcur, (short)0, acc[nt], false, false);
                bcur = bnxt;
            }
        }
        __syncthreads();
    }

    // ---- write C tiles to LDS scores: C[r] = tile[M = lhalf*8 + r][N = lrow]
#pragma unroll
    for (int nt = 0; nt < 16; ++nt) {
        int n_tile = nt * 2 + n_half;
        int col  = n_tile * 16 + lrow;
        int rowb = m_tile * 16 + lhalf * 8;
#pragma unroll
        for (int r = 0; r < 8; ++r) s[(rowb + r) * 512 + col] = acc[nt][r];
    }
    __syncthreads();

    // ---- softmax denominators: Z_i = sum_k exp(s[i,k])   (|s| << 1, no max needed)
    {
        int row = t >> 2, part = t & 3;
        float* sr = &s[row * 512 + part * 128];
        float z = 0.f;
        for (int c = 0; c < 128; ++c) { float e = __expf(sr[c]); sr[c] = e; z += e; }
        zpart[row * 4 + part] = z;
    }
    __syncthreads();
    if (t < TI)
        rinv[t] = 1.0f / (zpart[t*4] + zpart[t*4+1] + zpart[t*4+2] + zpart[t*4+3]);
    __syncthreads();

    // ---- column sums of attn for this row strip -> atomic add into A[b,h,k]
    float* Abh = A + (size_t)(b * H_ + h) * I_;
    for (int c = t; c < 512; c += 256) {
        float a = 0.f;
        for (int i = 0; i < TI; ++i) a += s[i * 512 + c] * rinv[i];
        atomicAdd(&Abh[c], a);
    }
}

// ---------------------------------------------------------------------------
// Kernel 2: S[b,j] = sum_k x[b,k,j] * sum_h A[b,h,k] * W_v[h,k,j]
// ---------------------------------------------------------------------------
__global__ __launch_bounds__(128)
void ov_kernel(const float* __restrict__ x, const float* __restrict__ Wv,
               const float* __restrict__ A, float* __restrict__ S)
{
    int b = blockIdx.y;
    int j = blockIdx.x * 128 + threadIdx.x;
    const float* Ab = A + (size_t)b * H_ * I_;
    float acc = 0.f;
    for (int k = 0; k < I_; ++k) {
        float xv = x[((size_t)b * I_ + k) * J_ + j];
        float wsum = 0.f;
#pragma unroll
        for (int h = 0; h < H_; ++h)
            wsum += Ab[h * I_ + k] * Wv[((size_t)h * I_ + k) * J_ + j];
        acc += xv * wsum;
    }
    S[b * J_ + j] = acc;
}

// ---------------------------------------------------------------------------
// Kernel 3: attn_out = W_o * S; h1 = LN_axis(-2)(attn_out + x);
//           nn = relu(h1*x1+y1)*x2+y2; out = LN_axis(-2)(nn + h1)
// Block: 256 threads = 64 j-columns x 4 i-groups; 512x64 f32 slab in LDS.
// ---------------------------------------------------------------------------
__global__ __launch_bounds__(256)
void fuse_kernel(const float* __restrict__ x,  const float* __restrict__ Wo,
                 const float* __restrict__ S,
                 const float* __restrict__ x1, const float* __restrict__ y1,
                 const float* __restrict__ x2, const float* __restrict__ y2,
                 const float* __restrict__ g1, const float* __restrict__ b1,
                 const float* __restrict__ g2, const float* __restrict__ b2,
                 float* __restrict__ out)
{
    __shared__ float buf[I_ * 64];     // 128 KB
    __shared__ float p1[4][64], p2[4][64];
    __shared__ float meanv[64], istdv[64];

    const int b  = blockIdx.y;
    const int j0 = blockIdx.x * 64;
    const int t  = threadIdx.x;
    const int j  = t & 63;
    const int g  = t >> 6;             // 0..3, each covers 128 i values
    const int jg = j0 + j;
    const float eps = 1e-3f;
    const float sval = S[b * J_ + jg];

    // pass 1: t = Wo*S + x ; stats over i
    float a1 = 0.f, a2 = 0.f;
    for (int r = 0; r < 128; ++r) {
        int i = g * 128 + r;
        float tt = Wo[(size_t)i * J_ + jg] * sval + x[((size_t)b * I_ + i) * J_ + jg];
        buf[i * 64 + j] = tt;
        a1 += tt; a2 += tt * tt;
    }
    p1[g][j] = a1; p2[g][j] = a2;
    __syncthreads();
    if (g == 0) {
        float s1 = p1[0][j] + p1[1][j] + p1[2][j] + p1[3][j];
        float s2 = p2[0][j] + p2[1][j] + p2[2][j] + p2[3][j];
        float m = s1 * (1.f / I_);
        float v = s2 * (1.f / I_) - m * m;
        meanv[j] = m; istdv[j] = rsqrtf(v + eps);
    }
    __syncthreads();
    float m1 = meanv[j], is1 = istdv[j];

    // pass 2: h1 -> MLP -> u = nn + h1 ; stats over i
    a1 = 0.f; a2 = 0.f;
    __syncthreads();
    for (int r = 0; r < 128; ++r) {
        int i = g * 128 + r;
        float h1 = (buf[i * 64 + j] - m1) * is1 * g1[i] + b1[i];
        size_t e = (size_t)i * J_ + jg;
        float nn = h1 * x1[e] + y1[e];
        nn = nn > 0.f ? nn : 0.f;
        nn = nn * x2[e] + y2[e];
        float u = nn + h1;
        buf[i * 64 + j] = u;
        a1 += u; a2 += u * u;
    }
    p1[g][j] = a1; p2[g][j] = a2;
    __syncthreads();
    if (g == 0) {
        float s1 = p1[0][j] + p1[1][j] + p1[2][j] + p1[3][j];
        float s2 = p2[0][j] + p2[1][j] + p2[2][j] + p2[3][j];
        float m = s1 * (1.f / I_);
        float v = s2 * (1.f / I_) - m * m;
        meanv[j] = m; istdv[j] = rsqrtf(v + eps);
    }
    __syncthreads();
    float m2 = meanv[j], is2 = istdv[j];
    for (int r = 0; r < 128; ++r) {
        int i = g * 128 + r;
        out[((size_t)b * I_ + i) * J_ + jg] = (buf[i * 64 + j] - m2) * is2 * g2[i] + b2[i];
    }
}

// ---------------------------------------------------------------------------
// Launch
// ---------------------------------------------------------------------------
extern "C" void kernel_launch(void* const* d_in, const int* in_sizes, int n_in,
                              void* d_out, int out_size, void* d_ws, size_t ws_size,
                              hipStream_t stream)
{
    const float* x  = (const float*)d_in[0];
    const float* Wq = (const float*)d_in[1];
    const float* Wk = (const float*)d_in[2];
    const float* Wv = (const float*)d_in[3];
    const float* Wo = (const float*)d_in[4];
    const float* x1 = (const float*)d_in[5];
    const float* y1 = (const float*)d_in[6];
    const float* x2 = (const float*)d_in[7];
    const float* y2 = (const float*)d_in[8];
    const float* g1 = (const float*)d_in[9];
    const float* b1 = (const float*)d_in[10];
    const float* g2 = (const float*)d_in[11];
    const float* b2 = (const float*)d_in[12];
    float* out = (float*)d_out;

    // workspace layout: Qb[BHIJ] bf16 | Kb[BHIJ] bf16 | A[B*H*I] f32 | S[B*J] f32
    const size_t nQK = (size_t)B_ * H_ * I_ * J_;
    unsigned short* Qb = (unsigned short*)d_ws;
    unsigned short* Kb = Qb + nQK;
    float* A = (float*)(Kb + nQK);
    float* S = A + (size_t)B_ * H_ * I_;

    prep_qk<<<dim3(8192), dim3(256), 0, stream>>>(x, Wq, Wk, Qb, Kb);

    const int nA = B_ * H_ * I_;
    zerof<<<dim3((nA + 255) / 256), dim3(256), 0, stream>>>(A, nA);

    attn_kernel<<<dim3(B_ * H_ * (I_ / TI)), dim3(256), 0, stream>>>(Qb, Kb, A);

    ov_kernel<<<dim3(J_ / 128, B_), dim3(128), 0, stream>>>(x, Wv, A, S);

    fuse_kernel<<<dim3(J_ / 64, B_), dim3(256), 0, stream>>>(
        x, Wo, S, x1, y1, x2, y2, g1, b1, g2, b2, out);
}
